// MultiHeadSelfAttention_22591527977187
// MI455X (gfx1250) — compile-verified, matrix operand order checked by reading, not run on hardware
//
#include <hip/hip_runtime.h>

#define D_MODEL 1024
#define SEQ     2048
#define BATCH   2
#define NHEAD   16
#define DKH     64
#define MTOT    (BATCH*SEQ)   // 4096 rows in the fused [B*S, D] view

typedef __bf16 bf16;
typedef __attribute__((ext_vector_type(16))) __bf16 v16bf;
typedef __attribute__((ext_vector_type(8)))  __bf16 v8bf;
typedef __attribute__((ext_vector_type(4)))  __bf16 v4bf;
typedef __attribute__((ext_vector_type(8)))  float  v8f;

union AFrag { v16bf v; v8bf h[2]; };

static __device__ __forceinline__ v8f vzero8() {
  v8f z;
#pragma unroll
  for (int i = 0; i < 8; ++i) z[i] = 0.0f;
  return z;
}

static __device__ __forceinline__ v8f wmma_bf16(const AFrag& a, const AFrag& b, v8f c) {
  return __builtin_amdgcn_wmma_f32_16x16x32_bf16(false, a.v, false, b.v,
                                                 (short)0, c, false, false);
}

// 16x32 bf16 fragment in the CDNA5 A-matrix layout:
//   lanes 0-15 : row = row0+lane,    dwords 0-3 = K kk+0..7,  dwords 4-7 = K kk+16..23
//   lanes 16-31: row = row0+lane-16, dwords 0-3 = K kk+8..15, dwords 4-7 = K kk+24..31
// Two 16-byte loads per lane. Works for LDS and global pointers.
static __device__ __forceinline__ AFrag load_frag(const bf16* __restrict__ base,
                                                  int row0, int ld, int kk, int lane) {
  AFrag f;
  const int half = lane >> 4;
  const bf16* p = base + (size_t)(row0 + (lane & 15)) * ld + kk + half * 8;
  f.h[0] = *(const v8bf*)(p);
  f.h[1] = *(const v8bf*)(p + 16);
  return f;
}

// ---------------------------------------------------------------------------
// GEMM: Out = A(bf16 or f32, [MTOT,1024]) x W^T(f32 [N=1024,K=1024]) + bias
// MODE 0: store bf16 scattered into [B,H,S,Dk] (QKV proj), value scaled
// MODE 1: store f32 row-major [MTOT,1024] (output proj)
// Block tile 256x128, BK=64. 8 waves in 4x2 grid; each wave 64x64 =
// 4x4 WMMA tiles -> 16 WMMAs per 8 fragment loads per k-substep.
// ---------------------------------------------------------------------------
template <int A_IS_BF16, int MODE>
__global__ __launch_bounds__(256)
void gemm_wmma(const void* __restrict__ Aptr,
               const float* __restrict__ W,
               const float* __restrict__ bias,
               void* __restrict__ Outp,
               float scale)
{
  __shared__ bf16 As[256 * 64];   // 32 KB
  __shared__ bf16 Bs[128 * 64];   // 16 KB

  const int tid  = threadIdx.x;
  const int lane = tid & 31;
  const int wid  = tid >> 5;
  const int wm   = wid >> 1;       // 0..3
  const int wn   = wid & 1;        // 0..1
  const int n0   = blockIdx.x * 128;
  const int m0   = blockIdx.y * 256;

  v8f acc[4][4];
#pragma unroll
  for (int tm = 0; tm < 4; ++tm)
#pragma unroll
    for (int tn = 0; tn < 4; ++tn) acc[tm][tn] = vzero8();

  for (int k0 = 0; k0 < D_MODEL; k0 += 64) {
    __syncthreads();
    // ---- A tile: 256x64 into LDS as bf16 ----
    if (A_IS_BF16) {
      const bf16* A = (const bf16*)Aptr;
#pragma unroll
      for (int p = 0; p < 8; ++p) {
        int e = p * 2048 + tid * 8;
        int r = e >> 6, c = e & 63;
        *(v8bf*)&As[r * 64 + c] =
            *(const v8bf*)(A + (size_t)(m0 + r) * D_MODEL + k0 + c);
      }
    } else {
      const float* A = (const float*)Aptr;
#pragma unroll
      for (int p = 0; p < 16; ++p) {
        int e = p * 1024 + tid * 4;
        int r = e >> 6, c = e & 63;
        float4 t = *(const float4*)(A + (size_t)(m0 + r) * D_MODEL + k0 + c);
        v4bf b;
        b[0] = (bf16)t.x; b[1] = (bf16)t.y; b[2] = (bf16)t.z; b[3] = (bf16)t.w;
        *(v4bf*)&As[r * 64 + c] = b;
      }
    }
    // ---- W tile: rows n0..n0+127, cols k0..k0+63 (row-major [N,K]) ----
#pragma unroll
    for (int p = 0; p < 8; ++p) {
      int e = p * 1024 + tid * 4;
      int r = e >> 6, c = e & 63;
      float4 t = *(const float4*)(W + (size_t)(n0 + r) * D_MODEL + k0 + c);
      v4bf b;
      b[0] = (bf16)t.x; b[1] = (bf16)t.y; b[2] = (bf16)t.z; b[3] = (bf16)t.w;
      *(v4bf*)&Bs[r * 64 + c] = b;
    }
    __syncthreads();

#pragma unroll
    for (int kk = 0; kk < 64; kk += 32) {
      AFrag af[4], bf[4];
#pragma unroll
      for (int tm = 0; tm < 4; ++tm)
        af[tm] = load_frag(As, wm * 64 + tm * 16, 64, kk, lane);
#pragma unroll
      for (int tn = 0; tn < 4; ++tn)
        bf[tn] = load_frag(Bs, wn * 64 + tn * 16, 64, kk, lane);
#pragma unroll
      for (int tm = 0; tm < 4; ++tm)
#pragma unroll
        for (int tn = 0; tn < 4; ++tn)
          acc[tm][tn] = wmma_bf16(af[tm], bf[tn], acc[tm][tn]);
    }
  }

  // ---- Epilogue ----
  const int half = lane >> 4, ll = lane & 15;
#pragma unroll
  for (int tm = 0; tm < 4; ++tm)
#pragma unroll
    for (int tn = 0; tn < 4; ++tn)
#pragma unroll
      for (int r = 0; r < 8; ++r) {
        int grow = m0 + wm * 64 + tm * 16 + r + half * 8;
        int gcol = n0 + wn * 64 + tn * 16 + ll;
        float v = acc[tm][tn][r] + bias[gcol];
        if (MODE == 0) {
          v *= scale;
          int b = grow >> 11, s = grow & (SEQ - 1);
          int h = gcol >> 6,  d = gcol & (DKH - 1);
          ((bf16*)Outp)[(((size_t)(b * NHEAD + h)) * SEQ + s) * DKH + d] = (bf16)v;
        } else {
          ((float*)Outp)[(size_t)grow * D_MODEL + gcol] = v;
        }
      }
}

// ---------------------------------------------------------------------------
// Flash attention: grid (S/128, B*H), 8 waves, each wave owns 16 query rows.
// 128-key tiles: K straight copy to LDS ([key][dk] = B layout for Q K^T);
// V transposed into LDS ([dk][key]) for P V. Online softmax in f32; P staged
// bf16 through a per-wave LDS strip. Next tile prefetched into L2/WGP$.
// ---------------------------------------------------------------------------
__global__ __launch_bounds__(256)
void flash_attn(const bf16* __restrict__ Q, const bf16* __restrict__ K,
                const bf16* __restrict__ V, bf16* __restrict__ Ctx)
{
  __shared__ bf16 Ks[128 * 64];      // 16 KB  [key][dk]
  __shared__ bf16 Vs[64 * 128];      // 16 KB  [dk][key]
  __shared__ bf16 Ps[8][16 * 128];   // 32 KB  per-wave P strips

  const int tid  = threadIdx.x;
  const int lane = tid & 31;
  const int wid  = tid >> 5;
  const int half = lane >> 4, ll = lane & 15;
  const int bh = blockIdx.y;
  const int b  = bh >> 4, h = bh & (NHEAD - 1);
  const size_t base = (size_t)bh * SEQ * DKH;
  const int q0 = blockIdx.x * 128 + wid * 16;

  AFrag qf[2];
#pragma unroll
  for (int kk = 0; kk < 2; ++kk)
    qf[kk] = load_frag(Q + base, q0, DKH, kk * 32, lane);

  float mi[8], li[8];
#pragma unroll
  for (int r = 0; r < 8; ++r) { mi[r] = -__builtin_inff(); li[r] = 0.0f; }
  v8f o[4];
#pragma unroll
  for (int tn = 0; tn < 4; ++tn) o[tn] = vzero8();

  for (int key0 = 0; key0 < SEQ; key0 += 128) {
    // prefetch next tile while we compute this one
    if (key0 + 128 < SEQ) {
      const bf16* kn = K + base + (size_t)(key0 + 128) * DKH + tid * 8;
      const bf16* vn = V + base + (size_t)(key0 + 128) * DKH + tid * 8;
#pragma unroll
      for (int p = 0; p < 4; ++p) {
        __builtin_prefetch(kn + p * 2048, 0, 3);
        __builtin_prefetch(vn + p * 2048, 0, 3);
      }
    }
    __syncthreads();
#pragma unroll
    for (int p = 0; p < 4; ++p) {
      int e = p * 2048 + tid * 8;
      int r = e >> 6, c = e & 63;
      *(v8bf*)&Ks[r * 64 + c] =
          *(const v8bf*)(K + base + (size_t)(key0 + r) * DKH + c);
      v8bf vv = *(const v8bf*)(V + base + (size_t)(key0 + r) * DKH + c);
#pragma unroll
      for (int j = 0; j < 8; ++j) Vs[(c + j) * 128 + r] = vv[j];
    }
    __syncthreads();

    // scores: 16 q-rows x 128 keys (Q already scaled by 1/sqrt(Dk))
    v8f sa[8];
#pragma unroll
    for (int tn = 0; tn < 8; ++tn) sa[tn] = vzero8();
#pragma unroll
    for (int kk = 0; kk < 2; ++kk)
#pragma unroll
      for (int tn = 0; tn < 8; ++tn) {
        AFrag kb = load_frag(Ks, tn * 16, 64, kk * 32, lane);
        sa[tn] = wmma_bf16(qf[kk], kb, sa[tn]);
      }

    // online softmax per row (rows live per (vgpr r, lane half-group))
#pragma unroll
    for (int r = 0; r < 8; ++r) {
      float mx = sa[0][r];
#pragma unroll
      for (int tn = 1; tn < 8; ++tn) mx = fmaxf(mx, sa[tn][r]);
#pragma unroll
      for (int off = 8; off >= 1; off >>= 1)
        mx = fmaxf(mx, __shfl_xor(mx, off, 32));
      float mnew  = fmaxf(mi[r], mx);
      float alpha = __expf(mi[r] - mnew);
      float rs = 0.0f;
#pragma unroll
      for (int tn = 0; tn < 8; ++tn) {
        float p = __expf(sa[tn][r] - mnew);
        sa[tn][r] = p;
        rs += p;
      }
#pragma unroll
      for (int off = 8; off >= 1; off >>= 1)
        rs += __shfl_xor(rs, off, 32);
      li[r] = li[r] * alpha + rs;
      mi[r] = mnew;
#pragma unroll
      for (int tn = 0; tn < 4; ++tn) o[tn][r] *= alpha;
    }

    // stage P as bf16 in per-wave LDS strip, reload in A-fragment layout
    bf16* pb = &Ps[wid][0];
#pragma unroll
    for (int tn = 0; tn < 8; ++tn)
#pragma unroll
      for (int r = 0; r < 8; ++r)
        pb[(r + half * 8) * 128 + tn * 16 + ll] = (bf16)sa[tn][r];
    __asm__ volatile("" ::: "memory");

#pragma unroll
    for (int kk = 0; kk < 4; ++kk) {
      AFrag pa = load_frag(pb, 0, 128, kk * 32, lane);
#pragma unroll
      for (int tn = 0; tn < 4; ++tn) {
        AFrag vb = load_frag(Vs, tn * 16, 128, kk * 32, lane);
        o[tn] = wmma_bf16(pa, vb, o[tn]);
      }
    }
  }

  // normalize and write ctx in [B,S,D] layout (bf16)
#pragma unroll
  for (int r = 0; r < 8; ++r) {
    float inv = 1.0f / li[r];
    int s = q0 + r + half * 8;
#pragma unroll
    for (int tn = 0; tn < 4; ++tn) {
      float v = o[tn][r] * inv;
      Ctx[((size_t)(b * SEQ) + s) * D_MODEL + h * DKH + tn * 16 + ll] = (bf16)v;
    }
  }
}

extern "C" void kernel_launch(void* const* d_in, const int* in_sizes, int n_in,
                              void* d_out, int out_size, void* d_ws, size_t ws_size,
                              hipStream_t stream) {
  const float* x  = (const float*)d_in[0];
  const float* wq = (const float*)d_in[1];
  const float* bq = (const float*)d_in[2];
  const float* wk = (const float*)d_in[3];
  const float* bk = (const float*)d_in[4];
  const float* wv = (const float*)d_in[5];
  const float* bv = (const float*)d_in[6];
  const float* wo = (const float*)d_in[7];
  const float* bo = (const float*)d_in[8];

  const size_t elems = (size_t)MTOT * D_MODEL;
  bf16* Q = (bf16*)d_ws;
  bf16* K = Q + elems;
  bf16* V = K + elems;
  bf16* C = V + elems;   // 4 x 8MB = 32MB workspace

  dim3 blk(256);
  dim3 gg(D_MODEL / 128, MTOT / 256);
  const float scale = 0.125f;  // 1/sqrt(Dk), folded into Q

  gemm_wmma<0, 0><<<gg, blk, 0, stream>>>(x, wq, bq, (void*)Q, scale);
  gemm_wmma<0, 0><<<gg, blk, 0, stream>>>(x, wk, bk, (void*)K, 1.0f);
  gemm_wmma<0, 0><<<gg, blk, 0, stream>>>(x, wv, bv, (void*)V, 1.0f);
  flash_attn<<<dim3(SEQ / 128, BATCH * NHEAD), blk, 0, stream>>>(Q, K, V, C);
  gemm_wmma<1, 1><<<gg, blk, 0, stream>>>(C, wo, bo, d_out, 1.0f);
}
